// PointNetPET_47029891891381
// MI455X (gfx1250) — matrix operand order repaired and checked by reading, not compiled, not run on hardware
//
#include <hip/hip_runtime.h>

// ---------------------------------------------------------------------------
// PointNet-PET for MI455X (gfx1250, wave32, WMMA, async global->LDS)
// ---------------------------------------------------------------------------
#define NPTS        1024
#define BATCH       32
#define T_DIM       64
#define H_DIM       96
#define W_DIM       192
#define FLAT_SZ     (T_DIM * 2 * H_DIM * W_DIM)     // 2,359,296 per event
#define EPS_BN      1e-5f
#define TWO_PI      6.283185307179586f

typedef __attribute__((ext_vector_type(16))) __bf16 v16bf;
typedef __attribute__((ext_vector_type(8)))  float  v8f;

union Vu { v16bf v; unsigned u[8]; };

// ---------------------------------------------------------------------------
// K0: f32 -> bf16 weight conversion
// ---------------------------------------------------------------------------
__global__ void cvt_bf16_kernel(const float* __restrict__ src,
                                __bf16* __restrict__ dst, int n) {
    int i = blockIdx.x * blockDim.x + threadIdx.x;
    if (i < n) dst[i] = (__bf16)src[i];
}

// ---------------------------------------------------------------------------
// K1: stable stream compaction (first NPTS fired voxels) + point features.
// One 1024-thread block per batch. wave32 ballot + cross-wave LDS scan.
// pts layout: [b][n][5] (x, y, z, t, intensity), f32.
// ---------------------------------------------------------------------------
__global__ __launch_bounds__(1024) void extract_points_kernel(
    const float* __restrict__ burst, float* __restrict__ pts) {
    const int b    = blockIdx.x;
    const int tid  = threadIdx.x;
    const int lane = tid & 31;
    const int wid  = tid >> 5;
    const float* flat = burst + (size_t)b * FLAT_SZ;

    __shared__ int s_wsum[32];
    __shared__ int s_idx[NPTS];

    int total = 0;  // uniform across block
    for (int base = 0; base < FLAT_SZ; base += 1024) {
        float val = flat[base + tid];
        bool pred = val > 1e-3f;
        unsigned mask = __builtin_amdgcn_ballot_w32(pred);
        if (lane == 0) s_wsum[wid] = __popc(mask);
        __syncthreads();
        int woff = 0;
        #pragma unroll
        for (int w = 0; w < 32; ++w) woff += (w < wid) ? s_wsum[w] : 0;
        int btot = 0;
        #pragma unroll
        for (int w = 0; w < 32; ++w) btot += s_wsum[w];
        int prefix = __popc(mask & ((1u << lane) - 1u));
        int pos = total + woff + prefix;
        if (pred && pos < NPTS) s_idx[pos] = base + tid;
        total += btot;
        __syncthreads();
        if (total >= NPTS) break;
    }
    __syncthreads();

    int ceff = total < NPTS ? total : NPTS;
    float x = 0.f, y = 0.f, z = 0.f, tm = 0.f, inten = 0.f;
    if (ceff > 0) {
        int sel = s_idx[(tid < ceff) ? tid : (tid % ceff)];
        int wi = sel % W_DIM;
        int hi = (sel / W_DIM) % H_DIM;
        int ci = (sel / (W_DIM * H_DIM)) % 2;
        int ti =  sel / (W_DIM * H_DIM * 2);
        float theta = (float)wi * (TWO_PI / (float)(W_DIM - 1));
        float r = (ci == 0) ? 400.0f : 450.0f;
        x  = r * __cosf(theta);
        y  = r * __sinf(theta);
        z  = -150.0f + (float)hi * (300.0f / (float)(H_DIM - 1));
        tm = (float)ti * (1.0f / (float)(T_DIM - 1));
        inten = flat[sel];
    }
    float* p = pts + ((size_t)b * NPTS + tid) * 5;
    p[0] = x; p[1] = y; p[2] = z; p[3] = tm; p[4] = inten;
}

// ---------------------------------------------------------------------------
// K2: conv1 (5 -> 64) + BN + ReLU, output bf16 point-major [b][n][64]
// ---------------------------------------------------------------------------
__global__ __launch_bounds__(256) void conv1_kernel(
    const float* __restrict__ pts,
    const float* __restrict__ w1, const float* __restrict__ cb,
    const float* __restrict__ g,  const float* __restrict__ bb,
    const float* __restrict__ bm, const float* __restrict__ bv,
    __bf16* __restrict__ x1) {
    __shared__ float sw[64 * 5];
    __shared__ float ss[64], sbias[64];
    for (int i = threadIdx.x; i < 320; i += blockDim.x) sw[i] = w1[i];
    for (int i = threadIdx.x; i < 64; i += blockDim.x) {
        float sc = g[i] * rsqrtf(bv[i] + EPS_BN);
        ss[i] = sc;
        sbias[i] = bb[i] + (cb[i] - bm[i]) * sc;
    }
    __syncthreads();

    int gi = blockIdx.x * blockDim.x + threadIdx.x;   // 0 .. 32767
    const float* p = pts + (size_t)gi * 5;
    float f0 = p[0], f1 = p[1], f2 = p[2], f3 = p[3], f4 = p[4];
    __bf16* out = x1 + (size_t)gi * 64;
    #pragma unroll 8
    for (int o = 0; o < 64; ++o) {
        const float* w = sw + o * 5;
        float acc = f0 * w[0] + f1 * w[1] + f2 * w[2] + f3 * w[3] + f4 * w[4];
        acc = fmaxf(acc * ss[o] + sbias[o], 0.f);
        out[o] = (__bf16)acc;
    }
}

// ---------------------------------------------------------------------------
// K3: conv2 (64 -> 128) via v_wmma_f32_16x16x32_bf16 + fused BN/ReLU.
// 1 wave per block; block computes a 128x16 output tile (8 M-tiles, K=64).
// A layout (16x32 bf16): lane = M row (mod 16); lanes>=16 cover K+16;
// VGPR e packs K = 2e, 2e+1. B layout (32x16): lane = N
// (mod 16); lanes>=16 cover K+16; VGPR j packs K = 2j, 2j+1.
// ---------------------------------------------------------------------------
__global__ __launch_bounds__(32) void conv2_wmma_kernel(
    const __bf16* __restrict__ w2bf, const __bf16* __restrict__ x1,
    const float* __restrict__ cb, const float* __restrict__ g,
    const float* __restrict__ bb, const float* __restrict__ bm,
    const float* __restrict__ bv, __bf16* __restrict__ x2) {
    const int lane = threadIdx.x;
    const int n0   = blockIdx.x * 16;
    const int b    = blockIdx.y;
    const int hi   = lane >> 4;
    const int l16  = lane & 15;

    // B tiles: 2 K-steps from x1[b][n][64]
    Vu bt[2];
    const __bf16* xrow = x1 + ((size_t)(b * NPTS) + (n0 + l16)) * 64;
    #pragma unroll
    for (int ks = 0; ks < 2; ++ks)
        #pragma unroll
        for (int j = 0; j < 8; ++j)
            bt[ks].u[j] = *(const unsigned*)(xrow + ks * 32 + hi * 16 + 2 * j);

    __bf16* orow = x2 + ((size_t)(b * NPTS) + (n0 + l16)) * 128;
    #pragma unroll
    for (int mt = 0; mt < 8; ++mt) {
        Vu a0, a1;
        const __bf16* wrow = w2bf + (size_t)(mt * 16 + l16) * 64;
        #pragma unroll
        for (int e = 0; e < 8; ++e) {
            int koff = 2 * e + hi * 16;
            a0.u[e] = *(const unsigned*)(wrow + koff);
            a1.u[e] = *(const unsigned*)(wrow + 32 + koff);
        }
        v8f c = {};
        c = __builtin_amdgcn_wmma_f32_16x16x32_bf16(false, a0.v, false, bt[0].v,
                                                    (short)0, c, false, false);
        c = __builtin_amdgcn_wmma_f32_16x16x32_bf16(false, a1.v, false, bt[1].v,
                                                    (short)0, c, false, false);
        // D layout: VGPR r -> M = r + 8*hi, N = lane (mod 16)
        #pragma unroll
        for (int r = 0; r < 8; ++r) {
            int o = mt * 16 + r + hi * 8;
            float sc = g[o] * rsqrtf(bv[o] + EPS_BN);
            float y  = c[r] * sc + (bb[o] + (cb[o] - bm[o]) * sc);
            orow[o] = (__bf16)fmaxf(y, 0.f);
        }
    }
}

// ---------------------------------------------------------------------------
// K4: conv3 (128 -> 1024) via WMMA + fused BN/ReLU + global max-pool over N.
// Grid: (chunk=8, batch=32); 8 waves/block, wave w owns M-tile w of the
// 128-channel chunk. x2 N-tiles staged into LDS with the CDNA5 async
// global->LDS path (GLOBAL_LOAD_ASYNC_TO_LDS_B128 + s_wait_asynccnt), which
// bypasses VGPRs entirely and completes before the WMMAs read the tile.
// ---------------------------------------------------------------------------
__global__ __launch_bounds__(256) void conv3_pool_kernel(
    const __bf16* __restrict__ w3bf, const __bf16* __restrict__ x2,
    const float* __restrict__ cb, const float* __restrict__ g,
    const float* __restrict__ bb, const float* __restrict__ bm,
    const float* __restrict__ bv, float* __restrict__ pooled) {
    const int chunk = blockIdx.x;       // 0..7  -> channels chunk*128..+127
    const int b     = blockIdx.y;       // 0..31
    const int tid   = threadIdx.x;
    const int lane  = tid & 31;
    const int wv    = tid >> 5;         // 0..7
    const int hi    = lane >> 4;
    const int l16   = lane & 15;

    __shared__ __attribute__((aligned(16))) __bf16 sx[16 * 128];  // 4 KB tile

    // A tiles (weights) for this wave's 16 output rows, K = 128 -> 4 steps
    Vu a[4];
    const int mrow = chunk * 128 + wv * 16 + l16;
    const __bf16* wrow = w3bf + (size_t)mrow * 128;
    #pragma unroll
    for (int ks = 0; ks < 4; ++ks)
        #pragma unroll
        for (int e = 0; e < 8; ++e)
            a[ks].u[e] = *(const unsigned*)(wrow + ks * 32 +
                                            2 * e + hi * 16);

    // Per-lane fused BN constants (channel fixed per (lane, r))
    float s8[8], o8[8];
    #pragma unroll
    for (int r = 0; r < 8; ++r) {
        int ch = chunk * 128 + wv * 16 + r + hi * 8;
        float sc = g[ch] * rsqrtf(bv[ch] + EPS_BN);
        s8[r] = sc;
        o8[r] = bb[ch] + (cb[ch] - bm[ch]) * sc;
    }

    float mx[8];
    #pragma unroll
    for (int r = 0; r < 8; ++r) mx[r] = -INFINITY;

    // Per-thread staging addresses (16B per thread, 256 threads = 4 KB tile)
    const int pnt = tid >> 4;            // point within tile: 0..15
    const int seg = (tid & 15) * 8;      // 8 bf16 = 16B segment within row
    const unsigned ldsoff =
        (unsigned)(size_t)(void*)&sx[pnt * 128 + seg];   // LDS byte offset
    const __bf16* xb = x2 + (size_t)b * NPTS * 128;

    for (int nt = 0; nt < 64; ++nt) {
        __syncthreads();   // previous tile fully consumed by all waves
        {
            unsigned long long ga = (unsigned long long)(const void*)
                (xb + (size_t)(nt * 16 + pnt) * 128 + seg);
            // CDNA5 async copy: memory -> LDS, tracked by ASYNCcnt
            asm volatile("global_load_async_to_lds_b128 %0, %1, off"
                         :: "v"(ldsoff), "v"(ga) : "memory");
        }
        // Drain this wave's async queue, then barrier publishes tile to WG
        asm volatile("s_wait_asynccnt 0x0" ::: "memory");
        __syncthreads();

        Vu bt[4];
        #pragma unroll
        for (int ks = 0; ks < 4; ++ks)
            #pragma unroll
            for (int j = 0; j < 8; ++j)
                bt[ks].u[j] = *(const unsigned*)(&sx[l16 * 128 + ks * 32 +
                                                    hi * 16 + 2 * j]);
        v8f c = {};
        c = __builtin_amdgcn_wmma_f32_16x16x32_bf16(false, a[0].v, false, bt[0].v,
                                                    (short)0, c, false, false);
        c = __builtin_amdgcn_wmma_f32_16x16x32_bf16(false, a[1].v, false, bt[1].v,
                                                    (short)0, c, false, false);
        c = __builtin_amdgcn_wmma_f32_16x16x32_bf16(false, a[2].v, false, bt[2].v,
                                                    (short)0, c, false, false);
        c = __builtin_amdgcn_wmma_f32_16x16x32_bf16(false, a[3].v, false, bt[3].v,
                                                    (short)0, c, false, false);
        #pragma unroll
        for (int r = 0; r < 8; ++r) {
            float y = fmaxf(c[r] * s8[r] + o8[r], 0.f);
            mx[r] = fmaxf(mx[r], y);
        }
    }

    // Max-reduce the 16 N-columns across each 16-lane half
    #pragma unroll
    for (int r = 0; r < 8; ++r) {
        float v = mx[r];
        #pragma unroll
        for (int off = 8; off >= 1; off >>= 1)
            v = fmaxf(v, __shfl_xor(v, off, 16));
        if (l16 == 0)
            pooled[(size_t)b * 1024 + chunk * 128 + wv * 16 + r + hi * 8] = v;
    }
}

// ---------------------------------------------------------------------------
// K5/K6: fully-connected + BN + ReLU (f32)
// ---------------------------------------------------------------------------
__global__ void fc_bn_relu_kernel(
    const float* __restrict__ in, const float* __restrict__ w,
    const float* __restrict__ cb, const float* __restrict__ g,
    const float* __restrict__ bb, const float* __restrict__ bm,
    const float* __restrict__ bv, float* __restrict__ out,
    int IN, int OUT) {
    int i = blockIdx.x * blockDim.x + threadIdx.x;
    if (i >= BATCH * OUT) return;
    int b = i / OUT, o = i % OUT;
    const float* x  = in + (size_t)b * IN;
    const float* wr = w + (size_t)o * IN;
    float acc = 0.f;
    for (int k = 0; k < IN; k += 4)
        acc += x[k] * wr[k] + x[k + 1] * wr[k + 1] +
               x[k + 2] * wr[k + 2] + x[k + 3] * wr[k + 3];
    float sc = g[o] * rsqrtf(bv[o] + EPS_BN);
    out[(size_t)b * OUT + o] = fmaxf((acc + cb[o] - bm[o]) * sc + bb[o], 0.f);
}

// K7: final fc (256 -> 7), no BN/ReLU
__global__ void fc3_kernel(const float* __restrict__ in,
                           const float* __restrict__ w,
                           const float* __restrict__ cb,
                           float* __restrict__ out) {
    int i = blockIdx.x * blockDim.x + threadIdx.x;
    if (i >= BATCH * 7) return;
    int b = i / 7, o = i % 7;
    const float* x  = in + (size_t)b * 256;
    const float* wr = w + (size_t)o * 256;
    float acc = 0.f;
    for (int k = 0; k < 256; k += 4)
        acc += x[k] * wr[k] + x[k + 1] * wr[k + 1] +
               x[k + 2] * wr[k + 2] + x[k + 3] * wr[k + 3];
    out[(size_t)b * 7 + o] = acc + cb[o];
}

// ---------------------------------------------------------------------------
// Host launch
// ---------------------------------------------------------------------------
extern "C" void kernel_launch(void* const* d_in, const int* in_sizes, int n_in,
                              void* d_out, int out_size, void* d_ws, size_t ws_size,
                              hipStream_t stream) {
    (void)in_sizes; (void)n_in; (void)out_size; (void)ws_size;

    const float* burst = (const float*)d_in[0];
    const float* c1w = (const float*)d_in[1];
    const float* c1b = (const float*)d_in[2];
    const float* bn1g = (const float*)d_in[3], *bn1b = (const float*)d_in[4];
    const float* bn1m = (const float*)d_in[5], *bn1v = (const float*)d_in[6];
    const float* c2w = (const float*)d_in[7];
    const float* c2b = (const float*)d_in[8];
    const float* bn2g = (const float*)d_in[9],  *bn2b = (const float*)d_in[10];
    const float* bn2m = (const float*)d_in[11], *bn2v = (const float*)d_in[12];
    const float* c3w = (const float*)d_in[13];
    const float* c3b = (const float*)d_in[14];
    const float* bn3g = (const float*)d_in[15], *bn3b = (const float*)d_in[16];
    const float* bn3m = (const float*)d_in[17], *bn3v = (const float*)d_in[18];
    const float* f1w = (const float*)d_in[19];
    const float* f1b = (const float*)d_in[20];
    const float* bf1g = (const float*)d_in[21], *bf1b = (const float*)d_in[22];
    const float* bf1m = (const float*)d_in[23], *bf1v = (const float*)d_in[24];
    const float* f2w = (const float*)d_in[25];
    const float* f2b = (const float*)d_in[26];
    const float* bf2g = (const float*)d_in[27], *bf2b = (const float*)d_in[28];
    const float* bf2m = (const float*)d_in[29], *bf2v = (const float*)d_in[30];
    const float* f3w = (const float*)d_in[31];
    const float* f3b = (const float*)d_in[32];
    float* out = (float*)d_out;

    // Workspace layout (all offsets 16B aligned)
    char* ws = (char*)d_ws;
    __bf16* w2bf  = (__bf16*)(ws + 0);                 // 128*64*2      = 16,384
    __bf16* w3bf  = (__bf16*)(ws + 16384);             // 1024*128*2    = 262,144
    float*  pts   = (float*) (ws + 278528);            // 32*1024*5*4   = 655,360
    __bf16* x1    = (__bf16*)(ws + 933888);            // 32*1024*64*2  = 4,194,304
    __bf16* x2    = (__bf16*)(ws + 5128192);           // 32*1024*128*2 = 8,388,608
    float*  pooled= (float*) (ws + 13516800);          // 32*1024*4     = 131,072
    float*  f1buf = (float*) (ws + 13647872);          // 32*512*4      = 65,536
    float*  f2buf = (float*) (ws + 13713408);          // 32*256*4      = 32,768

    cvt_bf16_kernel<<<(128 * 64 + 255) / 256, 256, 0, stream>>>(c2w, w2bf, 128 * 64);
    cvt_bf16_kernel<<<(1024 * 128 + 255) / 256, 256, 0, stream>>>(c3w, w3bf, 1024 * 128);

    extract_points_kernel<<<BATCH, 1024, 0, stream>>>(burst, pts);

    conv1_kernel<<<(BATCH * NPTS) / 256, 256, 0, stream>>>(
        pts, c1w, c1b, bn1g, bn1b, bn1m, bn1v, x1);

    conv2_wmma_kernel<<<dim3(NPTS / 16, BATCH), 32, 0, stream>>>(
        w2bf, x1, c2b, bn2g, bn2b, bn2m, bn2v, x2);

    conv3_pool_kernel<<<dim3(8, BATCH), 256, 0, stream>>>(
        w3bf, x2, c3b, bn3g, bn3b, bn3m, bn3v, pooled);

    fc_bn_relu_kernel<<<(BATCH * 512 + 255) / 256, 256, 0, stream>>>(
        pooled, f1w, f1b, bf1g, bf1b, bf1m, bf1v, f1buf, 1024, 512);

    fc_bn_relu_kernel<<<(BATCH * 256 + 255) / 256, 256, 0, stream>>>(
        f1buf, f2w, f2b, bf2g, bf2b, bf2m, bf2v, f2buf, 512, 256);

    fc3_kernel<<<1, 256, 0, stream>>>(f2buf, f3w, f3b, out);
}
